// RelativePositionHead_21354577396445
// MI455X (gfx1250) — matrix operand order repaired and checked by reading, not compiled, not run on hardware
//
#include <hip/hip_runtime.h>

// RelativePositionHead for MI455X (gfx1250), wave32 + WMMA bf16.
// Phase 0: weights fp32 -> bf16 (once; removes per-WMMA conversion VALU).
// Phase 1: Q/K/V projections, software-pipelined bf16 WMMA, V stored transposed.
// Phase 2: flash attention, K/V loads hoisted for latency overlap.

#define NB 4        // batch
#define TT 4096     // sequence length
#define DM 1024     // d_model
#define DH 64       // d_head
#define MAXREL 8

typedef __attribute__((ext_vector_type(16))) __bf16          v16bf;
typedef __attribute__((ext_vector_type(8)))  float           v8f;
typedef __attribute__((ext_vector_type(4)))  float           f32x4;
typedef __attribute__((ext_vector_type(8)))  unsigned short  u16x8;
typedef __attribute__((ext_vector_type(16))) unsigned short  u16x16;
typedef __attribute__((ext_vector_type(2)))  unsigned        u32x2;
typedef __attribute__((ext_vector_type(4)))  unsigned        u32x4;

static __device__ __forceinline__ unsigned short f2bf(float f) {
    unsigned u = __builtin_bit_cast(unsigned, f);
    u += 0x7FFFu + ((u >> 16) & 1u);     // round-to-nearest-even
    return (unsigned short)(u >> 16);
}

// packed fp32x2 -> bf16x2 (one v_cvt_pk_bf16_f32 when available)
#if __has_builtin(__builtin_amdgcn_cvt_pk_bf16_f32)
static __device__ __forceinline__ unsigned f2bf2(float lo, float hi) {
    typedef __attribute__((ext_vector_type(2))) __bf16 v2bf;
    v2bf r = __builtin_amdgcn_cvt_pk_bf16_f32(lo, hi);
    return __builtin_bit_cast(unsigned, r);
}
#else
static __device__ __forceinline__ unsigned f2bf2(float lo, float hi) {
    return (unsigned)f2bf(lo) | ((unsigned)f2bf(hi) << 16);
}
#endif

static __device__ __forceinline__ v8f wmma_bf16(v16bf a, v16bf b, v8f c) {
    // D(f32 16x16) = A(bf16 16x32) * B(bf16 32x16) + C
    return __builtin_amdgcn_wmma_f32_16x16x32_bf16(
        false, a, false, b, (short)0, c, false, false);
}

union AFrag { v16bf bf; u16x8 h[2]; };

// ---------------------------------------------------------------------------
// Phase 0: convert Wq/Wk/Wv (fp32 [64,1024]) to bf16 workspace [3][64][1024]
// ---------------------------------------------------------------------------
__global__ __launch_bounds__(256)
void cvt_w_kernel(const float* __restrict__ Wq,
                  const float* __restrict__ Wk,
                  const float* __restrict__ Wv,
                  unsigned short* __restrict__ wbf) {
    const float* W = (blockIdx.y == 0) ? Wq : (blockIdx.y == 1) ? Wk : Wv;
    unsigned short* dst = wbf + (size_t)blockIdx.y * DH * DM;
    const int idx = blockIdx.x * 256 + threadIdx.x;      // 4 floats per thread
    f32x4 v = *(const f32x4*)(W + (size_t)idx * 4);
    u32x2 p;
    p.x = f2bf2(v.x, v.y);
    p.y = f2bf2(v.z, v.w);
    *(u32x2*)(dst + (size_t)idx * 4) = p;
}

// ---------------------------------------------------------------------------
// Phase 1: q = x@Wq^T, k = x@Wk^T, v = x@Wv^T  (per 16-row tile of x)
//   qbf,kbf: [B,T,64] bf16 ; vtbf: [B,64,T] bf16 (transposed)
// ks-outer / matrix-inner: 1 LDS A-load feeds 3 WMMAs; double-buffered loads.
// ---------------------------------------------------------------------------
__global__ __launch_bounds__(128)
void proj_qkv_kernel(const float* __restrict__ x,
                     const unsigned short* __restrict__ wbf,
                     unsigned short* __restrict__ qbf,
                     unsigned short* __restrict__ kbf,
                     unsigned short* __restrict__ vtbf) {
    __shared__ unsigned short xs[16 * DM];   // 32 KB bf16 x-tile

    const int tid = threadIdx.x;
    const int bt  = blockIdx.x;              // 0 .. B*T/16-1
    const int b   = bt / (TT / 16);
    const int t0  = (bt % (TT / 16)) * 16;

    // cooperative load of x tile, fp32 -> bf16 (packed converts)
    const float* xrow = x + (size_t)(b * TT + t0) * DM;
    for (int idx = tid; idx < (16 * DM) / 4; idx += 128) {
        f32x4 v = *(const f32x4*)(xrow + idx * 4);
        u32x2 p;
        p.x = f2bf2(v.x, v.y);
        p.y = f2bf2(v.z, v.w);
        *(u32x2*)(xs + idx * 4) = p;
    }
    __syncthreads();

    const int wave = tid >> 5;               // n-tile: head dims [16w,16w+16)
    const int lane = tid & 31;
    const int lm   = lane & 15;
    const int lh   = lane >> 4;

    const unsigned short* xsrow = xs + lm * DM + 16 * lh;               // A base
    const unsigned short* wrow  =
        wbf + (size_t)(wave * 16 + lm) * DM + 16 * lh;                  // B base

    v8f acc[3];
#pragma unroll
    for (int m = 0; m < 3; ++m)
        acc[m] = (v8f){0.f, 0.f, 0.f, 0.f, 0.f, 0.f, 0.f, 0.f};

    // double-buffered fragments: issue ks+1 loads before ks WMMAs
    AFrag  a[2];
    u16x16 bw[2][3];
    a[0].h[0] = *(const u16x8*)(xsrow);
    a[0].h[1] = *(const u16x8*)(xsrow + 8);
#pragma unroll
    for (int m = 0; m < 3; ++m)
        bw[0][m] = *(const u16x16*)(wrow + (size_t)m * DH * DM);

#pragma unroll
    for (int ks = 0; ks < DM / 32; ++ks) {
        const int cur = ks & 1, nxt = cur ^ 1;
        if (ks + 1 < DM / 32) {
            const unsigned short* ap = xsrow + (ks + 1) * 32;
            a[nxt].h[0] = *(const u16x8*)(ap);
            a[nxt].h[1] = *(const u16x8*)(ap + 8);
#pragma unroll
            for (int m = 0; m < 3; ++m)
                bw[nxt][m] =
                    *(const u16x16*)(wrow + (size_t)m * DH * DM + (ks + 1) * 32);
        }
#pragma unroll
        for (int m = 0; m < 3; ++m)
            acc[m] = wmma_bf16(a[cur].bf,
                               __builtin_bit_cast(v16bf, bw[cur][m]), acc[m]);
    }

    // stores: C layout -> lane holds rows r+8h at column 16w+lm
#pragma unroll
    for (int m = 0; m < 2; ++m) {            // q, k row-major
        unsigned short* dst = (m == 0) ? qbf : kbf;
#pragma unroll
        for (int r = 0; r < 8; ++r) {
            const int t = t0 + r + 8 * lh;
            dst[(size_t)(b * TT + t) * DH + wave * 16 + lm] = f2bf(acc[m][r]);
        }
    }
    {                                        // v transposed: contiguous in t
        u32x4 w;
#pragma unroll
        for (int r = 0; r < 4; ++r) w[r] = f2bf2(acc[2][2 * r], acc[2][2 * r + 1]);
        unsigned short* dst =
            vtbf + ((size_t)(b * DH + wave * 16 + lm)) * TT + t0 + 8 * lh;
        *(u32x4*)dst = w;
    }
}

// ---------------------------------------------------------------------------
// Phase 2: causal flash attention with relative-position bias.
// 1 wave per 16-query tile, 4 waves per block.
// K loads issued first, V loads second (retire in order), so S-WMMAs wait only
// for K while softmax VALU hides the V-load latency before the P.V WMMAs.
// ---------------------------------------------------------------------------
__global__ __launch_bounds__(128)
void attn_kernel(const unsigned short* __restrict__ qbf,
                 const unsigned short* __restrict__ kbf,
                 const unsigned short* __restrict__ vtbf,
                 const float* __restrict__ rel_emb,
                 float* __restrict__ out) {
    __shared__ unsigned short pTile[4][16 * 32];   // per-wave P transpose buffer
    __shared__ float relTab[2 * MAXREL + 1];

    const int tid = threadIdx.x;
    if (tid < 2 * MAXREL + 1) relTab[tid] = rel_emb[tid];
    __syncthreads();

    const int wave = tid >> 5;
    const int lane = tid & 31;
    const int ln   = lane & 15;
    const int lh   = lane >> 4;

    const int qt = blockIdx.x * 4 + wave;          // query-tile id
    const int b  = qt / (TT / 16);
    const int t0 = (qt % (TT / 16)) * 16;
    unsigned short* pw = &pTile[wave][0];

    // Q A-fragments for the two K=32 d-steps (resident in VGPRs)
    AFrag aQ[2];
    {
        const unsigned short* qp = qbf + (size_t)(b * TT + t0 + ln) * DH;
#pragma unroll
        for (int ks = 0; ks < 2; ++ks) {
            aQ[ks].h[0] = *(const u16x8*)(qp + ks * 32 + 16 * lh);
            aQ[ks].h[1] = *(const u16x8*)(qp + ks * 32 + 8 + 16 * lh);
        }
    }

    float mrow[8], lsum[8];
    v8f o[4];
#pragma unroll
    for (int r = 0; r < 8; ++r) { mrow[r] = -3.0e38f; lsum[r] = 0.f; }
#pragma unroll
    for (int nt = 0; nt < 4; ++nt)
        o[nt] = (v8f){0.f, 0.f, 0.f, 0.f, 0.f, 0.f, 0.f, 0.f};

    const float scale = 0.125f;                    // 1/sqrt(64)
    const float LOG2E = 1.4426950408889634f;

    const unsigned short* krow = kbf + (size_t)(b * TT + ln) * DH + 16 * lh;
    const unsigned short* vrow = vtbf + (size_t)(b * DH + ln) * TT + 16 * lh;

    for (int jc = 0; jc < t0 + 16; jc += 32) {     // 32-key chunks up to diagonal
        // ---- prefetch next chunk's K/V (global_prefetch_b8) ----------------
        const int jn = jc + 32;
        if (jn < t0 + 16) {
            __builtin_prefetch(krow + (size_t)jn * DH, 0, 1);
            __builtin_prefetch(krow + (size_t)(jn + 16) * DH, 0, 1);
#pragma unroll
            for (int nt = 0; nt < 4; ++nt)
                __builtin_prefetch(vrow + (size_t)(nt * 16) * TT + jn, 0, 1);
        }

        // ---- issue all chunk loads: K first, then V ------------------------
        u16x16 bhK[2][2];                          // [key-tile][d-step]
#pragma unroll
        for (int kt = 0; kt < 2; ++kt)
#pragma unroll
            for (int ks = 0; ks < 2; ++ks)
                bhK[kt][ks] = *(const u16x16*)(
                    krow + (size_t)(jc + kt * 16) * DH + ks * 32);
        u16x16 bhV[4];                             // [head-dim tile]
#pragma unroll
        for (int nt = 0; nt < 4; ++nt)
            bhV[nt] = *(const u16x16*)(vrow + (size_t)(nt * 16) * TT + jc);

        // ---- S = Q K^T for two 16-key tiles --------------------------------
        v8f s[2];
#pragma unroll
        for (int kt = 0; kt < 2; ++kt) {
            v8f acc = {0.f, 0.f, 0.f, 0.f, 0.f, 0.f, 0.f, 0.f};
#pragma unroll
            for (int ks = 0; ks < 2; ++ks)
                acc = wmma_bf16(aQ[ks].bf,
                                __builtin_bit_cast(v16bf, bhK[kt][ks]), acc);
            s[kt] = acc;
        }
        // ---- scale + rel bias + causal mask, per-lane tile max -------------
        float tmax[8];
#pragma unroll
        for (int r = 0; r < 8; ++r) {
            const int i = t0 + r + 8 * lh;
            int j0 = jc + ln;
            int d0 = j0 - i; d0 = d0 < -MAXREL ? -MAXREL : (d0 > MAXREL ? MAXREL : d0);
            float v0 = (j0 <= i) ? (s[0][r] * scale + relTab[d0 + MAXREL]) : -3.0e38f;
            int j1 = jc + 16 + ln;
            int d1 = j1 - i; d1 = d1 < -MAXREL ? -MAXREL : (d1 > MAXREL ? MAXREL : d1);
            float v1 = (j1 <= i) ? (s[1][r] * scale + relTab[d1 + MAXREL]) : -3.0e38f;
            s[0][r] = v0; s[1][r] = v1;
            tmax[r] = fmaxf(v0, v1);
        }
        // row max across the 16-lane column group (C-layout half)
#pragma unroll
        for (int off = 1; off < 16; off <<= 1)
#pragma unroll
            for (int r = 0; r < 8; ++r)
                tmax[r] = fmaxf(tmax[r], __shfl_xor(tmax[r], off, 32));

        // ---- online softmax update ----------------------------------------
        float alpha[8], rsum[8];
#pragma unroll
        for (int r = 0; r < 8; ++r) {
            const float mnew = fmaxf(mrow[r], tmax[r]);
            alpha[r] = exp2f((mrow[r] - mnew) * LOG2E);
            mrow[r]  = mnew;
            const float p0 = exp2f((s[0][r] - mnew) * LOG2E);
            const float p1 = exp2f((s[1][r] - mnew) * LOG2E);
            rsum[r] = p0 + p1;
            pw[(r + 8 * lh) * 32 + ln]      = f2bf(p0);   // P -> LDS (row,col)
            pw[(r + 8 * lh) * 32 + 16 + ln] = f2bf(p1);
        }
#pragma unroll
        for (int off = 1; off < 16; off <<= 1)
#pragma unroll
            for (int r = 0; r < 8; ++r)
                rsum[r] += __shfl_xor(rsum[r], off, 32);
#pragma unroll
        for (int r = 0; r < 8; ++r) lsum[r] = lsum[r] * alpha[r] + rsum[r];
#pragma unroll
        for (int nt = 0; nt < 4; ++nt)
#pragma unroll
            for (int r = 0; r < 8; ++r) o[nt][r] *= alpha[r];

        // cross-lane LDS dependency (C-layout stores -> A-layout loads):
        // DS pipe is in-order per wave, but force issue-order + no reordering.
        asm volatile("s_wait_dscnt 0" ::: "memory");

        // ---- P (A-layout) from LDS, then O += P V --------------------------
        AFrag aP;
        {
            const unsigned short* pp = pw + ln * 32 + 16 * lh;
            aP.h[0] = *(const u16x8*)(pp);
            aP.h[1] = *(const u16x8*)(pp + 8);
        }
#pragma unroll
        for (int nt = 0; nt < 4; ++nt)
            o[nt] = wmma_bf16(aP.bf, __builtin_bit_cast(v16bf, bhV[nt]), o[nt]);
    }

    // ---- epilogue: out = O / l ------------------------------------------
#pragma unroll
    for (int r = 0; r < 8; ++r) {
        const float inv = 1.0f / lsum[r];
        const int t = t0 + r + 8 * lh;
        float* op = out + (size_t)(b * TT + t) * DH;
#pragma unroll
        for (int nt = 0; nt < 4; ++nt)
            op[nt * 16 + ln] = o[nt][r] * inv;
    }
}

// ---------------------------------------------------------------------------
extern "C" void kernel_launch(void* const* d_in, const int* in_sizes, int n_in,
                              void* d_out, int out_size, void* d_ws, size_t ws_size,
                              hipStream_t stream) {
    const float* x   = (const float*)d_in[0];
    const float* Wq  = (const float*)d_in[1];
    const float* Wk  = (const float*)d_in[2];
    const float* Wv  = (const float*)d_in[3];
    const float* rel = (const float*)d_in[4];
    float* out = (float*)d_out;

    // workspace: q, k (row-major) + v (transposed) + bf16 weights -> ~6.4 MB
    unsigned short* qbf  = (unsigned short*)d_ws;
    unsigned short* kbf  = qbf + (size_t)NB * TT * DH;
    unsigned short* vtbf = kbf + (size_t)NB * TT * DH;
    unsigned short* wbf  = vtbf + (size_t)NB * TT * DH;

    dim3 wgrid(DH * DM / (256 * 4), 3);
    cvt_w_kernel<<<wgrid, 256, 0, stream>>>(Wq, Wk, Wv, wbf);
    proj_qkv_kernel<<<NB * TT / 16, 128, 0, stream>>>(x, wbf, qbf, kbf, vtbf);
    attn_kernel<<<NB * TT / 16 / 4, 128, 0, stream>>>(qbf, kbf, vtbf, rel, out);
}
